// EncoderLayer_23613730194051
// MI455X (gfx1250) — matrix operand order repaired and checked by reading, not compiled, not run on hardware
//
#include <hip/hip_runtime.h>
#include <hip/hip_bf16.h>

typedef __attribute__((ext_vector_type(16))) __bf16 v16bf;
typedef __attribute__((ext_vector_type(8)))  __bf16 v8bf;
typedef __attribute__((ext_vector_type(4)))  __bf16 v4bf;
typedef __attribute__((ext_vector_type(8)))  float  v8f;
typedef int v4i __attribute__((vector_size(4 * sizeof(int))));

// ---- CDNA5 async global->LDS path (compile-safe feature detection) ----
#if defined(__has_builtin)
#if __has_builtin(__builtin_amdgcn_global_load_async_to_lds_b128)
#define USE_ASYNC_COPY 1
#endif
#if __has_builtin(__builtin_amdgcn_s_wait_asynccnt)
#define HAVE_WAIT_ASYNC_BUILTIN 1
#endif
#endif
#ifndef USE_ASYNC_COPY
#define USE_ASYNC_COPY 0
#endif
#ifndef HAVE_WAIT_ASYNC_BUILTIN
#define HAVE_WAIT_ASYNC_BUILTIN 0
#endif

namespace {

constexpr int   kD    = 1024;
constexpr int   kDff  = 4096;
constexpr int   kB    = 4;
constexpr int   kS    = 2048;
constexpr int   kRows = kB * kS;   // 8192
constexpr float kEps   = 1e-6f;
constexpr float kSlope = 0.01f;

constexpr int BM = 128, BN = 128, BK = 32;
constexpr int LDSS = 40;                 // padded row stride (80B, 16B-aligned)
constexpr int TILE_ELEMS = BM * LDSS;    // one matrix tile in LDS
constexpr int BUF_ELEMS  = 2 * TILE_ELEMS;  // A tile + B tile

__device__ __forceinline__ float wave_sum(float v) {
#pragma unroll
  for (int o = 16; o > 0; o >>= 1) v += __shfl_xor(v, o, 32);
  return v;
}
__device__ __forceinline__ float wave_max(float v) {
#pragma unroll
  for (int o = 16; o > 0; o >>= 1) v = fmaxf(v, __shfl_xor(v, o, 32));
  return v;
}

#if USE_ASYNC_COPY
__device__ __forceinline__ void wait_async0() {
#if HAVE_WAIT_ASYNC_BUILTIN
  __builtin_amdgcn_s_wait_asynccnt(0);
#else
  asm volatile("s_wait_asynccnt 0x0" ::: "memory");
#endif
}
__device__ __forceinline__ void async_b128(const __bf16* g, __bf16* l) {
  __builtin_amdgcn_global_load_async_to_lds_b128(
      (__attribute__((address_space(1))) v4i*)(void*)const_cast<__bf16*>(g),
      (__attribute__((address_space(3))) v4i*)(void*)l, 0, 0);
}
#endif

// ---------------- elementwise f32 -> bf16 ----------------
__global__ __launch_bounds__(256)
void cvt_f32_to_bf16(const float* __restrict__ in, __bf16* __restrict__ out, long long n) {
  long long i = ((long long)blockIdx.x * 256 + threadIdx.x) * 4;
  if (i >= n) return;
  float4 f = *reinterpret_cast<const float4*>(in + i);
  v4bf o;
  o[0] = (__bf16)f.x; o[1] = (__bf16)f.y; o[2] = (__bf16)f.z; o[3] = (__bf16)f.w;
  *reinterpret_cast<v4bf*>(out + i) = o;
}

// ---------------- transpose f32 [R,C] -> bf16 [C,R] ----------------
__global__ __launch_bounds__(256)
void transpose_f32_to_bf16(const float* __restrict__ in, __bf16* __restrict__ out,
                           int R, int C) {
  __shared__ float tile[32][33];
  const int c0 = blockIdx.x * 32, r0 = blockIdx.y * 32;
  const int tx = threadIdx.x, ty = threadIdx.y;
#pragma unroll
  for (int k = 0; k < 32; k += 8)
    tile[ty + k][tx] = in[(long long)(r0 + ty + k) * C + c0 + tx];
  __syncthreads();
#pragma unroll
  for (int k = 0; k < 32; k += 8)
    out[(long long)(c0 + ty + k) * R + r0 + tx] = (__bf16)tile[tx][ty + k];
}

// ---------------- batched transpose bf16 [R,C] -> bf16 [C,R] ----------------
__global__ __launch_bounds__(256)
void transpose_bf16(const __bf16* __restrict__ in, __bf16* __restrict__ out,
                    int R, int C) {
  __shared__ __bf16 tile[32][33];
  in  += (long long)blockIdx.z * R * C;
  out += (long long)blockIdx.z * R * C;
  const int c0 = blockIdx.x * 32, r0 = blockIdx.y * 32;
  const int tx = threadIdx.x, ty = threadIdx.y;
#pragma unroll
  for (int k = 0; k < 32; k += 8)
    tile[ty + k][tx] = in[(long long)(r0 + ty + k) * C + c0 + tx];
  __syncthreads();
#pragma unroll
  for (int k = 0; k < 32; k += 8)
    out[(long long)(c0 + ty + k) * R + r0 + tx] = tile[tx][ty + k];
}

// ---------------- bf16 WMMA GEMM:  C[m,n] = scale * sum_k A[m,k]*Bt[n,k] + bias[n] ----------------
// A: [M,K] bf16 row-major; Bt: [N,K] bf16 row-major (B pre-transposed).
// Double-buffered LDS, one barrier per K-step, async global->LDS when available.
template <bool OUT_BF16, bool LRELU>
__global__ __launch_bounds__(256)
void gemm_bf16_wmma(const __bf16* __restrict__ A, const __bf16* __restrict__ Bt,
                    const float* __restrict__ bias,
                    float* __restrict__ Cf, __bf16* __restrict__ Cb,
                    int M, int N, int K,
                    long long sA, long long sB, long long sC, float scale) {
  __shared__ __align__(16) __bf16 lds[2 * BUF_ELEMS];

  const int tid    = threadIdx.x;
  const int wave   = tid >> 5;
  const int lane   = tid & 31;
  const int lane16 = lane & 15;
  const int half   = lane >> 4;

  const int waveM = wave >> 2;    // 0..1
  const int waveN = wave & 3;     // 0..3

  A  += (long long)blockIdx.z * sA;
  Bt += (long long)blockIdx.z * sB;
  const long long cOff = (long long)blockIdx.z * sC;

  const int rowBase = blockIdx.y * BM;
  const int colBase = blockIdx.x * BN;

  v8f acc[4][2];
#pragma unroll
  for (int i = 0; i < 4; ++i)
#pragma unroll
    for (int j = 0; j < 2; ++j)
#pragma unroll
      for (int e = 0; e < 8; ++e) acc[i][j][e] = 0.0f;

  // staging: 256 threads x (2 A-chunks + 2 B-chunks) of 16B each
  const int sr = tid >> 2;          // 0..63
  const int sc = (tid & 3) * 8;     // 0,8,16,24
  const int ldsOff = sr * LDSS + sc;

  const __bf16* aPtr0 = A  + (long long)(rowBase + sr)      * K + sc;
  const __bf16* aPtr1 = A  + (long long)(rowBase + sr + 64) * K + sc;
  const __bf16* bPtr0 = Bt + (long long)(colBase + sr)      * K + sc;
  const __bf16* bPtr1 = Bt + (long long)(colBase + sr + 64) * K + sc;

  const int nt = K / BK;

  // compute one K-step from LDS buffer `base`
  auto compute_tile = [&](const __bf16* base) {
    const __bf16* bA = base;
    const __bf16* bB = base + TILE_ELEMS;
    v16bf af[4];
#pragma unroll
    for (int i = 0; i < 4; ++i) {
      const int r  = waveM * 64 + i * 16 + lane16;
      const int ko = half * 8;
      v8bf lo = *reinterpret_cast<const v8bf*>(&bA[r * LDSS + ko]);
      v8bf hi = *reinterpret_cast<const v8bf*>(&bA[r * LDSS + ko + 16]);
#pragma unroll
      for (int e = 0; e < 8; ++e) { af[i][e] = lo[e]; af[i][e + 8] = hi[e]; }
    }
    v16bf bfr[2];
#pragma unroll
    for (int j = 0; j < 2; ++j) {
      const int n  = waveN * 32 + j * 16 + lane16;
      const int ko = half * 8;
      v8bf lo = *reinterpret_cast<const v8bf*>(&bB[n * LDSS + ko]);
      v8bf hi = *reinterpret_cast<const v8bf*>(&bB[n * LDSS + ko + 16]);
#pragma unroll
      for (int e = 0; e < 8; ++e) { bfr[j][e] = lo[e]; bfr[j][e + 8] = hi[e]; }
    }
#pragma unroll
    for (int i = 0; i < 4; ++i)
#pragma unroll
      for (int j = 0; j < 2; ++j)
        acc[i][j] = __builtin_amdgcn_wmma_f32_16x16x32_bf16(
            false, af[i], false, bfr[j], (short)0, acc[i][j], false, false);
  };

#if USE_ASYNC_COPY
  // ---- async double-buffered pipeline ----
  auto issue = [&](int k0, int buf) {
    __bf16* d = lds + buf * BUF_ELEMS;
    async_b128(aPtr0 + k0, d + ldsOff);
    async_b128(aPtr1 + k0, d + 64 * LDSS + ldsOff);
    async_b128(bPtr0 + k0, d + TILE_ELEMS + ldsOff);
    async_b128(bPtr1 + k0, d + TILE_ELEMS + 64 * LDSS + ldsOff);
  };
  issue(0, 0);
  for (int kt = 0; kt < nt; ++kt) {
    wait_async0();        // own tile-kt loads have landed in LDS
    __syncthreads();      // everyone's have
    if (kt + 1 < nt) issue((kt + 1) * BK, (kt + 1) & 1);  // overlaps with compute
    compute_tile(lds + (kt & 1) * BUF_ELEMS);
  }
#else
  // ---- register-prefetch double-buffered pipeline ----
  v8bf ra0 = *reinterpret_cast<const v8bf*>(aPtr0);
  v8bf ra1 = *reinterpret_cast<const v8bf*>(aPtr1);
  v8bf rb0 = *reinterpret_cast<const v8bf*>(bPtr0);
  v8bf rb1 = *reinterpret_cast<const v8bf*>(bPtr1);
  for (int kt = 0; kt < nt; ++kt) {
    __bf16* d = lds + (kt & 1) * BUF_ELEMS;
    *reinterpret_cast<v8bf*>(&d[ldsOff])                           = ra0;
    *reinterpret_cast<v8bf*>(&d[64 * LDSS + ldsOff])               = ra1;
    *reinterpret_cast<v8bf*>(&d[TILE_ELEMS + ldsOff])              = rb0;
    *reinterpret_cast<v8bf*>(&d[TILE_ELEMS + 64 * LDSS + ldsOff])  = rb1;
    __syncthreads();
    if (kt + 1 < nt) {  // issue next tile's loads before compute (latency hidden)
      const int k0 = (kt + 1) * BK;
      ra0 = *reinterpret_cast<const v8bf*>(aPtr0 + k0);
      ra1 = *reinterpret_cast<const v8bf*>(aPtr1 + k0);
      rb0 = *reinterpret_cast<const v8bf*>(bPtr0 + k0);
      rb1 = *reinterpret_cast<const v8bf*>(bPtr1 + k0);
    }
    compute_tile(d);
  }
#endif

  // epilogue
#pragma unroll
  for (int j = 0; j < 2; ++j) {
    const int n   = colBase + waveN * 32 + j * 16 + lane16;
    const float bv = bias ? bias[n] : 0.0f;
#pragma unroll
    for (int i = 0; i < 4; ++i) {
      const int m0 = rowBase + waveM * 64 + i * 16 + half * 8;
#pragma unroll
      for (int e = 0; e < 8; ++e) {
        float x = acc[i][j][e] * scale + bv;
        if (LRELU) x = (x > 0.0f) ? x : kSlope * x;
        const long long idx = cOff + (long long)(m0 + e) * N + n;
        if (OUT_BF16) Cb[idx] = (__bf16)x;
        else          Cf[idx] = x;
      }
    }
  }
}

// ---------------- row softmax over S=2048, then + intensity, store bf16 ----------------
__global__ __launch_bounds__(256)
void softmax_addbias(const float* __restrict__ scores, const float* __restrict__ inten,
                     __bf16* __restrict__ attn) {
  __shared__ float redA[8];
  __shared__ float redB[8];
  const long long row = blockIdx.x;
  const float* s = scores + row * kS;
  const float* b = inten  + row * kS;
  __bf16*      o = attn   + row * kS;
  const int tid = threadIdx.x;
  const int wave = tid >> 5, lane = tid & 31;

  float v[8];
  float m = -1e30f;
#pragma unroll
  for (int e = 0; e < 8; ++e) { v[e] = s[tid + e * 256]; m = fmaxf(m, v[e]); }
  m = wave_max(m);
  if (lane == 0) redA[wave] = m;
  __syncthreads();
  m = redA[0];
#pragma unroll
  for (int i = 1; i < 8; ++i) m = fmaxf(m, redA[i]);

  float sum = 0.0f;
#pragma unroll
  for (int e = 0; e < 8; ++e) { v[e] = __expf(v[e] - m); sum += v[e]; }
  sum = wave_sum(sum);
  if (lane == 0) redB[wave] = sum;
  __syncthreads();
  sum = 0.0f;
#pragma unroll
  for (int i = 0; i < 8; ++i) sum += redB[i];
  const float r = 1.0f / sum;
#pragma unroll
  for (int e = 0; e < 8; ++e)
    o[tid + e * 256] = (__bf16)(v[e] * r + b[tid + e * 256]);
}

// ---------------- fused residual add + LayerNorm over D=1024 ----------------
template <bool WRITE_BF16>
__global__ __launch_bounds__(256)
void add_layernorm(const float* __restrict__ a, const float* __restrict__ res,
                   const float* __restrict__ g, const float* __restrict__ be,
                   float* __restrict__ outF, __bf16* __restrict__ outB) {
  __shared__ float red1[8], red2[8];
  const long long base = (long long)blockIdx.x * kD;
  const int tid = threadIdx.x;
  const int wave = tid >> 5, lane = tid & 31;

  float x[4];
  float s1 = 0.0f, s2 = 0.0f;
#pragma unroll
  for (int e = 0; e < 4; ++e) {
    const int c = tid + e * 256;
    x[e] = a[base + c] + res[base + c];
    s1 += x[e];
    s2 += x[e] * x[e];
  }
  s1 = wave_sum(s1);
  s2 = wave_sum(s2);
  if (lane == 0) { red1[wave] = s1; red2[wave] = s2; }
  __syncthreads();
  s1 = 0.0f; s2 = 0.0f;
#pragma unroll
  for (int i = 0; i < 8; ++i) { s1 += red1[i]; s2 += red2[i]; }
  const float mu  = s1 * (1.0f / kD);
  const float var = s2 * (1.0f / kD) - mu * mu;
  const float inv = rsqrtf(var + kEps);
#pragma unroll
  for (int e = 0; e < 4; ++e) {
    const int c = tid + e * 256;
    const float y = (x[e] - mu) * inv * g[c] + be[c];
    outF[base + c] = y;
    if (WRITE_BF16) outB[base + c] = (__bf16)y;
  }
}

}  // namespace

extern "C" void kernel_launch(void* const* d_in, const int* in_sizes, int n_in,
                              void* d_out, int out_size, void* d_ws, size_t ws_size,
                              hipStream_t stream) {
  (void)in_sizes; (void)n_in; (void)out_size; (void)ws_size;
  const float* X     = (const float*)d_in[0];
  const float* inten = (const float*)d_in[1];
  const float* Wq = (const float*)d_in[2];
  const float* bq = (const float*)d_in[3];
  const float* Wk = (const float*)d_in[4];
  const float* bk = (const float*)d_in[5];
  const float* Wv = (const float*)d_in[6];
  const float* bvv = (const float*)d_in[7];
  const float* Wo = (const float*)d_in[8];
  const float* bo = (const float*)d_in[9];
  const float* W1 = (const float*)d_in[10];
  const float* b1 = (const float*)d_in[11];
  const float* W2 = (const float*)d_in[12];
  const float* b2 = (const float*)d_in[13];
  const float* g1  = (const float*)d_in[14];
  const float* be1 = (const float*)d_in[15];
  const float* g2  = (const float*)d_in[16];
  const float* be2 = (const float*)d_in[17];
  float* out = (float*)d_out;

  char* ws = (char*)d_ws;
  size_t off = 0;
  auto alloc = [&](size_t bytes) -> char* {
    char* p = ws + off;
    off = (off + bytes + 255) & ~(size_t)255;
    return p;
  };

  __bf16* Xbf = (__bf16*)alloc((size_t)kRows * kD * 2);
  __bf16* Wqt = (__bf16*)alloc((size_t)kD * kD * 2);
  __bf16* Wkt = (__bf16*)alloc((size_t)kD * kD * 2);
  __bf16* Wvt = (__bf16*)alloc((size_t)kD * kD * 2);
  __bf16* Wot = (__bf16*)alloc((size_t)kD * kD * 2);
  __bf16* W1t = (__bf16*)alloc((size_t)kD * kDff * 2);
  __bf16* W2t = (__bf16*)alloc((size_t)kDff * kD * 2);
  __bf16* qbf = (__bf16*)alloc((size_t)kRows * kD * 2);
  __bf16* kbf = (__bf16*)alloc((size_t)kRows * kD * 2);
  __bf16* vbf = (__bf16*)alloc((size_t)kRows * kD * 2);
  __bf16* vT  = (__bf16*)alloc((size_t)kB * kD * kS * 2);
  float*  scores = (float*)alloc((size_t)kB * kS * kS * 4);   // 64MB (later reused as f1)
  __bf16* attnb  = (__bf16*)alloc((size_t)kB * kS * kS * 2);  // 32MB
  float*  pout   = (float*)alloc((size_t)kRows * kD * 4);     // 32MB (later reused as ffout)
  float*  hf     = (float*)alloc((size_t)kRows * kD * 4);     // 32MB
  // lifetime-checked aliases:
  __bf16* hbf   = qbf;              // q dead after scores GEMM
  __bf16* attnV = kbf;              // k dead after scores GEMM
  __bf16* f1    = (__bf16*)scores;  // scores dead after softmax
  float*  ffout = pout;             // proj-out dead after LN1

  const dim3 b256(256);
  const dim3 tb(32, 8);

  // 1. X -> bf16
  cvt_f32_to_bf16<<<(kRows * kD) / 1024, b256, 0, stream>>>(X, Xbf, (long long)kRows * kD);

  // 2. pre-transpose weights to [N,K] bf16
  transpose_f32_to_bf16<<<dim3(kD / 32, kD / 32), tb, 0, stream>>>(Wq, Wqt, kD, kD);
  transpose_f32_to_bf16<<<dim3(kD / 32, kD / 32), tb, 0, stream>>>(Wk, Wkt, kD, kD);
  transpose_f32_to_bf16<<<dim3(kD / 32, kD / 32), tb, 0, stream>>>(Wv, Wvt, kD, kD);
  transpose_f32_to_bf16<<<dim3(kD / 32, kD / 32), tb, 0, stream>>>(Wo, Wot, kD, kD);
  transpose_f32_to_bf16<<<dim3(kDff / 32, kD / 32), tb, 0, stream>>>(W1, W1t, kD, kDff);
  transpose_f32_to_bf16<<<dim3(kD / 32, kDff / 32), tb, 0, stream>>>(W2, W2t, kDff, kD);

  // 3. q = X@Wq + bq, k = X@Wk + bk, v = X@Wv + bv   (bf16 out)
  const dim3 gq(kD / BN, kRows / BM, 1);
  gemm_bf16_wmma<true, false><<<gq, b256, 0, stream>>>(
      Xbf, Wqt, bq, nullptr, qbf, kRows, kD, kD, 0, 0, 0, 1.0f);
  gemm_bf16_wmma<true, false><<<gq, b256, 0, stream>>>(
      Xbf, Wkt, bk, nullptr, kbf, kRows, kD, kD, 0, 0, 0, 1.0f);
  gemm_bf16_wmma<true, false><<<gq, b256, 0, stream>>>(
      Xbf, Wvt, bvv, nullptr, vbf, kRows, kD, kD, 0, 0, 0, 1.0f);

  // 4. per-batch v^T : [S,D] -> [D,S]
  transpose_bf16<<<dim3(kD / 32, kS / 32, kB), tb, 0, stream>>>(vbf, vT, kS, kD);

  // 5. scores = (q @ k^T) / sqrt(D)   (f32 out, batched)
  const dim3 gs(kS / BN, kS / BM, kB);
  gemm_bf16_wmma<false, false><<<gs, b256, 0, stream>>>(
      qbf, kbf, nullptr, scores, nullptr, kS, kS, kD,
      (long long)kS * kD, (long long)kS * kD, (long long)kS * kS, 0.03125f);

  // 6. attn = softmax(scores) + intensity   (bf16 out)
  softmax_addbias<<<kRows, b256, 0, stream>>>(scores, inten, attnb);

  // 7. attnV = attn @ v   (bf16 out, batched; B operand = vT)
  const dim3 ga(kD / BN, kS / BM, kB);
  gemm_bf16_wmma<true, false><<<ga, b256, 0, stream>>>(
      attnb, vT, nullptr, nullptr, attnV, kS, kD, kS,
      (long long)kS * kS, (long long)kD * kS, (long long)kS * kD, 1.0f);

  // 8. proj = attnV @ Wo + bo   (f32 out)
  gemm_bf16_wmma<false, false><<<gq, b256, 0, stream>>>(
      attnV, Wot, bo, pout, nullptr, kRows, kD, kD, 0, 0, 0, 1.0f);

  // 9. h = LayerNorm(proj + X)  -> f32 + bf16
  add_layernorm<true><<<kRows, b256, 0, stream>>>(pout, X, g1, be1, hf, hbf);

  // 10. f1 = leaky_relu(h @ W1 + b1)   (bf16 out)
  const dim3 gf1(kDff / BN, kRows / BM, 1);
  gemm_bf16_wmma<true, true><<<gf1, b256, 0, stream>>>(
      hbf, W1t, b1, nullptr, f1, kRows, kDff, kD, 0, 0, 0, 1.0f);

  // 11. ffn = f1 @ W2 + b2   (f32 out)
  gemm_bf16_wmma<false, false><<<gq, b256, 0, stream>>>(
      f1, W2t, b2, ffout, nullptr, kRows, kD, kDff, 0, 0, 0, 1.0f);

  // 12. out = LayerNorm(ffn + h)
  add_layernorm<false><<<kRows, b256, 0, stream>>>(ffout, hf, g2, be2, out, nullptr);
}
